// HeteroGraphEncoder_29901562314976
// MI455X (gfx1250) — compile-verified, matrix operand order checked by reading, not run on hardware
//
#include <hip/hip_runtime.h>
#include <hip/hip_bf16.h>
#include <stdint.h>
#include <stddef.h>

// ---------- sizes (match reference) ----------
#define N_EVENT 100000
#define N_TRACE 20000
#define E_EE    1000000
#define E_ET    500000
#define E_TE    500000
#define DIM     128
#define OUTD    64
#define NLAYERS 3
#define NB      64

// ---------- vector types ----------
typedef __attribute__((ext_vector_type(16))) _Float16 v16h;
typedef __attribute__((ext_vector_type(8)))  float    v8f;
typedef __attribute__((ext_vector_type(4)))  unsigned int u32x4;
typedef __attribute__((ext_vector_type(4)))  float    f32x4;
typedef __attribute__((ext_vector_type(4)))  _Float16 h16x4;

union FragU { v16h v; u32x4 q[2]; };

// LDS A-tile row stride (halves): +8 pad = 16B, keeps 16B-aligned fragment loads
#define LDS_STRIDE (DIM + 8)

// ============================================================
// utility kernels
// ============================================================
__global__ void zero4_kernel(f32x4* __restrict__ p, int n4) {
    int i = blockIdx.x * blockDim.x + threadIdx.x;
    if (i < n4) { f32x4 z = {0.f, 0.f, 0.f, 0.f}; p[i] = z; }
}

__global__ void cvt_f16_kernel(const float* __restrict__ in, _Float16* __restrict__ out, int n) {
    int i = blockIdx.x * blockDim.x + threadIdx.x;
    if (i < n) out[i] = (_Float16)in[i];
}

// out[i] = (f16)(a[i] + b[i])   -- fold Wr0 + Wr2 for the event root term
__global__ void sum_cvt_f16_kernel(const float* __restrict__ a, const float* __restrict__ b,
                                   _Float16* __restrict__ out, int n) {
    int i = blockIdx.x * blockDim.x + threadIdx.x;
    if (i < n) out[i] = (_Float16)(a[i] + b[i]);
}

// x[i,:] = emb[ids[i],:]   (one wave per node, float4 per lane)
__global__ void gather_rows_kernel(const float* __restrict__ emb, const int* __restrict__ ids,
                                   float* __restrict__ x, int n) {
    int gid = blockIdx.x * blockDim.x + threadIdx.x;
    int i = gid >> 5, lane = gid & 31;
    if (i >= n) return;
    int s = ids[i];
    *(f32x4*)(x + (size_t)i * DIM + lane * 4) =
        *(const f32x4*)(emb + (size_t)s * DIM + lane * 4);
}

// msum[dst,:] += x[src,:] ; cnt[dst] += 1   (one wave per edge)
__global__ void edge_scatter_kernel(const int* __restrict__ src, const int* __restrict__ dst,
                                    const float* __restrict__ x,
                                    float* __restrict__ msum, float* __restrict__ cnt, int E) {
    int gid = blockIdx.x * blockDim.x + threadIdx.x;
    int e = gid >> 5, lane = gid & 31;
    if (e >= E) return;
    int s = src[e], d = dst[e];
    f32x4 v = *(const f32x4*)(x + (size_t)s * DIM + lane * 4);
    float* o = msum + (size_t)d * DIM + lane * 4;
    atomicAdd(o + 0, v[0]);
    atomicAdd(o + 1, v[1]);
    atomicAdd(o + 2, v[2]);
    atomicAdd(o + 3, v[3]);
    if (lane == 0) atomicAdd(cnt + d, 1.0f);
}

// pooled[batch[i],:] += x[i,:]
__global__ void pool_add_kernel(const float* __restrict__ x, const int* __restrict__ batch,
                                float* __restrict__ pooled, int n) {
    int gid = blockIdx.x * blockDim.x + threadIdx.x;
    int i = gid >> 5, lane = gid & 31;
    if (i >= n) return;
    int b = batch[i];
    f32x4 v = *(const f32x4*)(x + (size_t)i * DIM + lane * 4);
    float* o = pooled + (size_t)b * DIM + lane * 4;
    atomicAdd(o + 0, v[0]);
    atomicAdd(o + 1, v[1]);
    atomicAdd(o + 2, v[2]);
    atomicAdd(o + 3, v[3]);
}

// ============================================================
// WMMA node update with fused mean-division:
//   out[16-row tile, 128 cols] = sum_t (A_t * rowscale_t) @ W_t^T  (+ biases)
// rowscale_t = 1/max(S_t[row],1) if S_t != null (mean terms), else 1.
// 256 threads = 8 waves; wave w computes column tile [w*16, w*16+16).
// A tiles: fp32 in global, scaled+converted to f16 in LDS (shared by waves).
// W tiles: pre-converted f16 in global, row-major [n][k] (i.e. W^T columns).
// Fragment layouts per CDNA5 ISA (16-bit A/B, f32 C/D).
// ============================================================
template <int NT>
__global__ __launch_bounds__(256) void node_update_kernel(
    const float* __restrict__ A0, const _Float16* __restrict__ W0, const float* __restrict__ S0,
    const float* __restrict__ A1, const _Float16* __restrict__ W1, const float* __restrict__ S1,
    const float* __restrict__ A2, const _Float16* __restrict__ W2, const float* __restrict__ S2,
    const float* __restrict__ biasA, const float* __restrict__ biasB,
    float* __restrict__ out) {
    __shared__ _Float16 ldsA[16 * LDS_STRIDE];

    const int row0 = blockIdx.x * 16;
    const int tid  = threadIdx.x;
    const int wave = tid >> 5;
    const int lane = tid & 31;
    const int m    = lane & 15;            // A row / D column-within-tile index
    const int kk   = (lane >> 4) * 8;      // per-lane K sub-offset (0 or 8)
    const int ncol = wave * 16 + m;        // global output column for B/D

    const float*    As[3] = {A0, A1, A2};
    const _Float16* Ws[3] = {W0, W1, W2};
    const float*    Ss[3] = {S0, S1, S2};

    v8f acc = {};
#pragma unroll
    for (int t = 0; t < NT; ++t) {
        const float* __restrict__ A = As[t];
        const _Float16* __restrict__ W = Ws[t];
        const float* __restrict__ S = Ss[t];

        __syncthreads();  // previous term's fragment reads done
        // stage 16x128 fp32 -> (scaled) f16: 512 float4 groups, 2 per thread
#pragma unroll
        for (int e4 = tid; e4 < (16 * DIM) / 4; e4 += 256) {
            int r  = e4 >> 5;              // 32 float4 per row
            int c4 = (e4 & 31) << 2;
            float sc = 1.0f;
            if (S) sc = 1.0f / fmaxf(S[row0 + r], 1.0f);
            f32x4 v = *(const f32x4*)(A + (size_t)(row0 + r) * DIM + c4);
            h16x4 h = {(_Float16)(v[0] * sc), (_Float16)(v[1] * sc),
                       (_Float16)(v[2] * sc), (_Float16)(v[3] * sc)};
            *(h16x4*)(ldsA + r * LDS_STRIDE + c4) = h;
        }
        __syncthreads();

#pragma unroll
        for (int kc = 0; kc < 4; ++kc) {   // K = 128 as 4 steps of 32
            FragU a, b;
            // A fragment: lane m holds row m; halves 0..7 = K kc*32+kk..+7,
            // halves 8..15 = K kc*32+16+kk..+23  (16B LDS loads)
            const _Float16* ap = ldsA + m * LDS_STRIDE + kc * 32 + kk;
            a.q[0] = *(const u32x4*)(ap);
            a.q[1] = *(const u32x4*)(ap + 16);
            // B fragment: lane holds column ncol of W^T = row ncol of W
            const _Float16* bp = W + (size_t)ncol * DIM + kc * 32 + kk;
            b.q[0] = *(const u32x4*)(bp);
            b.q[1] = *(const u32x4*)(bp + 16);
            acc = __builtin_amdgcn_wmma_f32_16x16x32_f16(
                false, a.v, false, b.v, (short)0, acc, false, false);
        }
    }

    float badd = 0.f;
    if (biasA) badd += biasA[ncol];
    if (biasB) badd += biasB[ncol];

    const int mb = (lane >> 4) * 8;        // C/D: VGPR r -> M = mb + r
#pragma unroll
    for (int r = 0; r < 8; ++r)
        out[(size_t)(row0 + mb + r) * DIM + ncol] = acc[r] + badd;
}

// ============================================================
// tiny MLP head (negligible cost, keep in fp32)
// ============================================================
__global__ void head1_kernel(const float* __restrict__ pooled, const float* __restrict__ W1,
                             const float* __restrict__ b1, float* __restrict__ h) {
    int r = blockIdx.x;      // 0..63
    int c = threadIdx.x;     // 0..127
    float acc = b1[c];
    const float* p = pooled + (size_t)r * DIM;
    const float* w = W1 + (size_t)c * DIM;
#pragma unroll 4
    for (int k = 0; k < DIM; ++k) acc += p[k] * w[k];
    h[(size_t)r * DIM + c] = fmaxf(acc, 0.f);
}

__global__ void head2_kernel(const float* __restrict__ h, const float* __restrict__ W2,
                             const float* __restrict__ b2, float* __restrict__ out) {
    int r = blockIdx.x;      // 0..63
    int c = threadIdx.x;     // 0..63
    float acc = b2[c];
    const float* p = h + (size_t)r * DIM;
    const float* w = W2 + (size_t)c * DIM;
#pragma unroll 4
    for (int k = 0; k < DIM; ++k) acc += p[k] * w[k];
    out[(size_t)r * OUTD + c] = acc;
}

// ============================================================
// host orchestration
// ============================================================
extern "C" void kernel_launch(void* const* d_in, const int* in_sizes, int n_in,
                              void* d_out, int out_size, void* d_ws, size_t ws_size,
                              hipStream_t stream) {
    const int*   event_ids   = (const int*)d_in[0];
    const int*   trace_ids   = (const int*)d_in[1];
    const int*   e2e_src     = (const int*)d_in[2];
    const int*   e2e_dst     = (const int*)d_in[3];
    const int*   e2t_src     = (const int*)d_in[4];
    const int*   e2t_dst     = (const int*)d_in[5];
    const int*   t2e_src     = (const int*)d_in[6];
    const int*   t2e_dst     = (const int*)d_in[7];
    const int*   event_batch = (const int*)d_in[8];
    const int*   trace_batch = (const int*)d_in[9];
    const float* emb_event   = (const float*)d_in[10];
    const float* emb_trace   = (const float*)d_in[11];
    const float* Wl          = (const float*)d_in[12];  // (3,3,128,128)
    const float* bl          = (const float*)d_in[13];  // (3,3,128)
    const float* Wr          = (const float*)d_in[14];  // (3,3,128,128)
    const float* W1          = (const float*)d_in[15];
    const float* b1          = (const float*)d_in[16];
    const float* W2          = (const float*)d_in[17];
    const float* b2          = (const float*)d_in[18];
    float* out = (float*)d_out;

    // ---- bump allocator on workspace ----
    uintptr_t base = (uintptr_t)d_ws;
    auto alloc = [&](size_t bytes) -> void* {
        uintptr_t p = (base + 255) & ~(uintptr_t)255;
        base = p + bytes;
        return (void*)p;
    };

    float* xE[2]; float* xT[2];
    xE[0] = (float*)alloc((size_t)N_EVENT * DIM * sizeof(float));
    xE[1] = (float*)alloc((size_t)N_EVENT * DIM * sizeof(float));
    xT[0] = (float*)alloc((size_t)N_TRACE * DIM * sizeof(float));
    xT[1] = (float*)alloc((size_t)N_TRACE * DIM * sizeof(float));
    // mean-sum buffers + counts laid out contiguously so one zero pass covers all
    float* mEE = (float*)alloc(((size_t)N_EVENT * DIM + (size_t)N_EVENT * DIM +
                                (size_t)N_TRACE * DIM + N_EVENT + N_EVENT + N_TRACE) *
                               sizeof(float));
    float* mTE = mEE + (size_t)N_EVENT * DIM;
    float* mET = mTE + (size_t)N_EVENT * DIM;
    float* cEE = mET + (size_t)N_TRACE * DIM;
    float* cTE = cEE + N_EVENT;
    float* cET = cTE + N_EVENT;
    const size_t meanRegion = (size_t)N_EVENT * DIM * 2 + (size_t)N_TRACE * DIM +
                              N_EVENT * 2 + N_TRACE;   // floats, all multiples of 4
    float* pooled = (float*)alloc((size_t)NB * DIM * sizeof(float));
    float* hbuf   = (float*)alloc((size_t)NB * DIM * sizeof(float));
    _Float16* Wl16 = (_Float16*)alloc((size_t)NLAYERS * 3 * DIM * DIM * sizeof(_Float16));
    _Float16* Wr16 = (_Float16*)alloc((size_t)NLAYERS * 3 * DIM * DIM * sizeof(_Float16));
    _Float16* WrS16 = (_Float16*)alloc((size_t)NLAYERS * DIM * DIM * sizeof(_Float16));

    const int nWElems = NLAYERS * 3 * DIM * DIM;  // 147456
    const int nMat    = DIM * DIM;                // 16384

    // ---- one-time-per-call weight conversion fp32 -> f16 ----
    cvt_f16_kernel<<<(nWElems + 255) / 256, 256, 0, stream>>>(Wl, Wl16, nWElems);
    cvt_f16_kernel<<<(nWElems + 255) / 256, 256, 0, stream>>>(Wr, Wr16, nWElems);
    // event root weight fold: WrS[l] = Wr[l,0] + Wr[l,2] (summed fp32, then f16)
    for (int l = 0; l < NLAYERS; ++l) {
        sum_cvt_f16_kernel<<<nMat / 256, 256, 0, stream>>>(
            Wr + (size_t)(l * 3 + 0) * nMat, Wr + (size_t)(l * 3 + 2) * nMat,
            WrS16 + (size_t)l * nMat, nMat);
    }

    // ---- embedding gather ----
    gather_rows_kernel<<<(N_EVENT * 32) / 256, 256, 0, stream>>>(emb_event, event_ids, xE[0], N_EVENT);
    gather_rows_kernel<<<(N_TRACE * 32) / 256, 256, 0, stream>>>(emb_trace, trace_ids, xT[0], N_TRACE);

    int cur = 0;
    for (int l = 0; l < NLAYERS; ++l) {
        int nxt = cur ^ 1;

        // zero mean sums + counts
        int n4 = (int)(meanRegion / 4);
        zero4_kernel<<<(n4 + 255) / 256, 256, 0, stream>>>((f32x4*)mEE, n4);

        // scatter each edge type (bandwidth-bound phase; b128 loads + f32 atomics)
        edge_scatter_kernel<<<((size_t)E_EE * 32) / 256, 256, 0, stream>>>(
            e2e_src, e2e_dst, xE[cur], mEE, cEE, E_EE);
        edge_scatter_kernel<<<((size_t)E_ET * 32) / 256, 256, 0, stream>>>(
            e2t_src, e2t_dst, xE[cur], mET, cET, E_ET);
        edge_scatter_kernel<<<((size_t)E_TE * 32) / 256, 256, 0, stream>>>(
            t2e_src, t2e_dst, xT[cur], mTE, cTE, E_TE);

        // dense updates (WMMA); mean division fused into A-tile staging
        const _Float16* Wl0 = Wl16 + (size_t)(l * 3 + 0) * nMat;
        const _Float16* Wl1 = Wl16 + (size_t)(l * 3 + 1) * nMat;
        const _Float16* Wl2 = Wl16 + (size_t)(l * 3 + 2) * nMat;
        const _Float16* Wr1 = Wr16 + (size_t)(l * 3 + 1) * nMat;
        const _Float16* WrS = WrS16 + (size_t)l * nMat;
        const float* bl0 = bl + (size_t)(l * 3 + 0) * DIM;
        const float* bl1 = bl + (size_t)(l * 3 + 1) * DIM;
        const float* bl2 = bl + (size_t)(l * 3 + 2) * DIM;

        // event dst: mean(mEE)@Wl0^T + mean(mTE)@Wl2^T + x_e@(Wr0+Wr2)^T + bl0 + bl2
        node_update_kernel<3><<<N_EVENT / 16, 256, 0, stream>>>(
            mEE, Wl0, cEE,
            mTE, Wl2, cTE,
            xE[cur], WrS, nullptr,
            bl0, bl2, xE[nxt]);
        // trace dst: mean(mET)@Wl1^T + x_t@Wr1^T + bl1
        node_update_kernel<2><<<N_TRACE / 16, 256, 0, stream>>>(
            mET, Wl1, cET,
            xT[cur], Wr1, nullptr,
            nullptr, nullptr, nullptr,
            bl1, nullptr, xT[nxt]);

        cur = nxt;
    }

    // ---- global add pool over both node types ----
    zero4_kernel<<<(NB * DIM / 4 + 255) / 256, 256, 0, stream>>>((f32x4*)pooled, NB * DIM / 4);
    pool_add_kernel<<<(N_EVENT * 32) / 256, 256, 0, stream>>>(xE[cur], event_batch, pooled, N_EVENT);
    pool_add_kernel<<<(N_TRACE * 32) / 256, 256, 0, stream>>>(xT[cur], trace_batch, pooled, N_TRACE);

    // ---- MLP head ----
    head1_kernel<<<NB, DIM, 0, stream>>>(pooled, W1, b1, hbuf);
    head2_kernel<<<NB, OUTD, 0, stream>>>(hbuf, W2, b2, out);
}
